// MPNN_56100862820630
// MI455X (gfx1250) — compile-verified, matrix operand order checked by reading, not run on hardware
//
#include <hip/hip_runtime.h>
#include <hip/hip_bf16.h>
#include <math.h>

// ---------------------------------------------------------------------------
// MPNN potential (REANN-style) for MI455X / gfx1250.
//
// Edge phase is memory/atomic bound -> coalesced float4 gathers +
// global_atomic_add_f32 scatter; per-edge transcendentals precomputed once.
// Atom MLPs (24->64->64->{8,1}) run as wave32 WMMA GEMMs
// (v_wmma_f32_16x16x32_f16), one 16-atom tile per wave, K padded to 32.
// silu uses v_exp_f32 + fast v_rcp_f32 (no IEEE divide chain).
// mp_params[1]'s MLP is dead code in the reference and is skipped.
// ---------------------------------------------------------------------------

#define NWAVE   8
#define R_MAX_L 3
#define NORBIT  24
#define NATOM   50000
#define NEIGH   1000000
#define CUTOFF  5.0f

typedef __attribute__((ext_vector_type(16))) _Float16 v16h;
typedef __attribute__((ext_vector_type(8)))  float    v8f;

// silu(x) = x * sigmoid(x), with fast hardware reciprocal (v_rcp_f32)
// instead of the IEEE divide expansion (div_scale + 2x Newton + fixup).
__device__ __forceinline__ float silu(float x) {
    return x * __builtin_amdgcn_rcpf(1.0f + __expf(-x));
}

// ---------------------------------------------------------------------------
// K1: per-atom embedding MLP 1 -> 16 -> 16 -> 24 (silu, silu, linear).
// Tiny K, bandwidth trivial -> plain VALU. Also seeds coefA = emb[:, :8].
// ---------------------------------------------------------------------------
__global__ __launch_bounds__(256)
void emb_kernel(const float* __restrict__ species,
                const float* __restrict__ W0, const float* __restrict__ B0,   // 1x16
                const float* __restrict__ W1, const float* __restrict__ B1,   // 16x16
                const float* __restrict__ W2, const float* __restrict__ B2,   // 16x24
                float* __restrict__ emb, float* __restrict__ coefA)
{
    int a = blockIdx.x * blockDim.x + threadIdx.x;
    if (a >= NATOM) return;
    float s = species[a];
    float h0[16], h1[16];
    #pragma unroll
    for (int j = 0; j < 16; ++j) h0[j] = silu(s * W0[j] + B0[j]);
    #pragma unroll
    for (int j = 0; j < 16; ++j) {
        float v = B1[j];
        #pragma unroll
        for (int k = 0; k < 16; ++k) v += h0[k] * W1[k * 16 + j];
        h1[j] = silu(v);
    }
    #pragma unroll
    for (int j = 0; j < 24; ++j) {
        float v = B2[j];
        #pragma unroll
        for (int k = 0; k < 16; ++k) v += h1[k] * W2[k * 24 + j];
        emb[a * 24 + j] = v;
        if (j < NWAVE) coefA[a * NWAVE + j] = v;   // coefficients (loop 0)
    }
}

// ---------------------------------------------------------------------------
// K2: per-edge constants (fixed across all density steps):
//   fc, sph[9](coor/CUTOFF), w[j] = fc * exp(-(alpha_j (d-center_j))^2)
// ---------------------------------------------------------------------------
__global__ __launch_bounds__(256)
void edge_pre_kernel(const float* __restrict__ cart,
                     const int*   __restrict__ aidx,
                     const float* __restrict__ shifts,
                     const float* __restrict__ emb,
                     float* __restrict__ fcE,
                     float* __restrict__ wE,
                     float* __restrict__ sphE)
{
    int e = blockIdx.x * blockDim.x + threadIdx.x;
    if (e >= NEIGH) return;
    int c = aidx[e];
    int n = aidx[NEIGH + e];
    float x = cart[0 * NATOM + n] - cart[0 * NATOM + c] + shifts[0 * NEIGH + e];
    float y = cart[1 * NATOM + n] - cart[1 * NATOM + c] + shifts[1 * NEIGH + e];
    float z = cart[2 * NATOM + n] - cart[2 * NATOM + c] + shifts[2 * NEIGH + e];
    float d = sqrtf(x * x + y * y + z * z);

    float u  = (cosf(d * (3.14159265358979323846f / CUTOFF)) + 1.0f) * 0.5f;
    float fc = u * u * u;
    fcE[e] = fc;

    float xs = x * (1.0f / CUTOFF), ys = y * (1.0f / CUTOFF), zs = z * (1.0f / CUTOFF);
    float r2 = xs * xs + ys * ys + zs * zs;
    const float c0  = 0.28209479177387814f;
    const float c1  = 0.4886025119029199f;
    const float c2a = 1.0925484305920792f;
    const float c2b = 0.31539156525252005f;
    const float c2c = 0.5462742152960396f;
    float sp[9];
    sp[0] = c0;
    sp[1] = c1 * ys;  sp[2] = c1 * zs;  sp[3] = c1 * xs;
    sp[4] = c2a * xs * ys;  sp[5] = c2a * ys * zs;
    sp[6] = c2b * (3.0f * zs * zs - r2);
    sp[7] = c2a * xs * zs;  sp[8] = c2c * (xs * xs - ys * ys);
    #pragma unroll
    for (int k = 0; k < 9; ++k) sphE[e * 9 + k] = sp[k];

    const float* ec = emb + n * 24;
    #pragma unroll
    for (int j = 0; j < NWAVE; ++j) {
        float t = ec[8 + j] * (d - ec[16 + j]);
        wE[e * NWAVE + j] = fc * __expf(-t * t);
    }
}

// ---------------------------------------------------------------------------
// K3: density step edge pass:
//   r[k][j] = prev[idx_n][k][j]*fc + sph[k] * (w[j]*coef[idx_n][j])
//   atomicAdd(next[idx_c][k][j], r)
// ---------------------------------------------------------------------------
__global__ __launch_bounds__(256)
void edge_scatter_kernel(const int*   __restrict__ aidx,
                         const float* __restrict__ fcE,
                         const float* __restrict__ wE,
                         const float* __restrict__ sphE,
                         const float* __restrict__ coefA,
                         const float* __restrict__ prev,   // [NATOM][9][8]
                         float*       __restrict__ next)   // [NATOM][9][8]
{
    int e = blockIdx.x * blockDim.x + threadIdx.x;
    if (e >= NEIGH) return;
    int c = aidx[e];
    int n = aidx[NEIGH + e];
    float fc = fcE[e];
    float wc[8];
    #pragma unroll
    for (int j = 0; j < 8; ++j) wc[j] = wE[e * 8 + j] * coefA[n * 8 + j];
    float sp[9];
    #pragma unroll
    for (int k = 0; k < 9; ++k) sp[k] = sphE[e * 9 + k];

    const float4* pv = reinterpret_cast<const float4*>(prev + (size_t)n * 72);
    float* nx = next + (size_t)c * 72;
    #pragma unroll
    for (int k = 0; k < 9; ++k) {
        float4 p0 = pv[k * 2 + 0];
        float4 p1 = pv[k * 2 + 1];
        float s = sp[k];
        atomicAdd(nx + k * 8 + 0, p0.x * fc + s * wc[0]);
        atomicAdd(nx + k * 8 + 1, p0.y * fc + s * wc[1]);
        atomicAdd(nx + k * 8 + 2, p0.z * fc + s * wc[2]);
        atomicAdd(nx + k * 8 + 3, p0.w * fc + s * wc[3]);
        atomicAdd(nx + k * 8 + 4, p1.x * fc + s * wc[4]);
        atomicAdd(nx + k * 8 + 5, p1.y * fc + s * wc[5]);
        atomicAdd(nx + k * 8 + 6, p1.z * fc + s * wc[6]);
        atomicAdd(nx + k * 8 + 7, p1.w * fc + s * wc[7]);
    }
}

// ---------------------------------------------------------------------------
// K4: density[a][INDEX_L[k]][j] += sum_sph[a][k][j]^2
// ---------------------------------------------------------------------------
__global__ __launch_bounds__(256)
void density_kernel(const float* __restrict__ ss,      // [NATOM][9][8]
                    float* __restrict__ density)       // [NATOM][3][8]
{
    int a = blockIdx.x * blockDim.x + threadIdx.x;
    if (a >= NATOM) return;
    const int IDXL[9] = {0, 1, 1, 1, 2, 2, 2, 2, 2};
    float dl[24];
    #pragma unroll
    for (int i = 0; i < 24; ++i) dl[i] = density[a * 24 + i];
    #pragma unroll
    for (int k = 0; k < 9; ++k) {
        int l = IDXL[k];
        #pragma unroll
        for (int j = 0; j < 8; ++j) {
            float v = ss[(size_t)a * 72 + k * 8 + j];
            dl[l * 8 + j] += v * v;
        }
    }
    #pragma unroll
    for (int i = 0; i < 24; ++i) density[a * 24 + i] = dl[i];
}

// ---------------------------------------------------------------------------
// K5/K6: WMMA MLP  X[NATOM][24] -> 64 (silu) -> 64 (silu) -> OUTD (linear).
// One wave32 per 16-atom tile (50000 = 3125 exact tiles), 4 waves/block.
// v_wmma_f32_16x16x32_f16; A/B f16 fragments built per the 16-bit 16x32
// layout (lanes 0-15: K 0..7/16..23, lanes 16-31: K 8..15/24..31).
// Hidden activations staged through LDS (per-wave slice, s_wait_dscnt).
// ---------------------------------------------------------------------------
template<int OUTD>
__global__ __launch_bounds__(128)
void mlp_wmma_kernel(const float* __restrict__ X,
                     const float* __restrict__ W0, const float* __restrict__ B0, // 24x64
                     const float* __restrict__ W1, const float* __restrict__ B1, // 64x64
                     const float* __restrict__ W2, const float* __restrict__ B2, // 64xOUTD
                     float* __restrict__ Y)                                      // [NATOM][OUTD]
{
    __shared__ _Float16 lds[4][16][80];   // per-wave 16x64 tile, padded stride

    const int lane = threadIdx.x & 31;
    const int wid  = threadIdx.x >> 5;
    const int tile = blockIdx.x * 4 + wid;
    if (tile * 16 >= NATOM) return;

    const int half = lane >> 4;      // 0: K 0..7/16..23, 1: K 8..15/24..31
    const int l15  = lane & 15;
    const int kb   = half * 8;
    const int arow = tile * 16 + l15;

    // ---- layer 0: A = X tile (K=24 padded to 32) --------------------------
    v16h a;
    #pragma unroll
    for (int i = 0; i < 16; ++i) {
        int k = kb + (i < 8 ? i : i + 8);
        a[i] = (k < 24) ? (_Float16)X[arow * 24 + k] : (_Float16)0.0f;
    }
    #pragma unroll
    for (int nt = 0; nt < 4; ++nt) {
        int col = nt * 16 + l15;
        v16h b;
        #pragma unroll
        for (int i = 0; i < 16; ++i) {
            int k = kb + (i < 8 ? i : i + 8);
            b[i] = (k < 24) ? (_Float16)W0[k * 64 + col] : (_Float16)0.0f;
        }
        v8f cacc = {};
        cacc = __builtin_amdgcn_wmma_f32_16x16x32_f16(false, a, false, b,
                                                      (short)0, cacc, false, false);
        float bias = B0[col];
        #pragma unroll
        for (int r = 0; r < 8; ++r) {
            float v = silu(cacc[r] + bias);
            lds[wid][r + 8 * half][nt * 16 + l15] = (_Float16)v;  // (M, N)
        }
    }
    asm volatile("s_wait_dscnt 0" ::: "memory");

    // ---- layer 1: 64x64, K = 2 chunks of 32 -------------------------------
    v16h a0, a1;
    #pragma unroll
    for (int i = 0; i < 16; ++i) {
        int k = kb + (i < 8 ? i : i + 8);
        a0[i] = lds[wid][l15][k];
        a1[i] = lds[wid][l15][32 + k];
    }
    v8f h2[4];
    #pragma unroll
    for (int nt = 0; nt < 4; ++nt) {
        int col = nt * 16 + l15;
        v16h b0, b1;
        #pragma unroll
        for (int i = 0; i < 16; ++i) {
            int k = kb + (i < 8 ? i : i + 8);
            b0[i] = (_Float16)W1[k * 64 + col];
            b1[i] = (_Float16)W1[(32 + k) * 64 + col];
        }
        v8f cacc = {};
        cacc = __builtin_amdgcn_wmma_f32_16x16x32_f16(false, a0, false, b0,
                                                      (short)0, cacc, false, false);
        cacc = __builtin_amdgcn_wmma_f32_16x16x32_f16(false, a1, false, b1,
                                                      (short)0, cacc, false, false);
        float bias = B1[col];
        #pragma unroll
        for (int r = 0; r < 8; ++r) h2[nt][r] = silu(cacc[r] + bias);
    }
    #pragma unroll
    for (int nt = 0; nt < 4; ++nt)
        #pragma unroll
        for (int r = 0; r < 8; ++r)
            lds[wid][r + 8 * half][nt * 16 + l15] = (_Float16)h2[nt][r];
    asm volatile("s_wait_dscnt 0" ::: "memory");

    // ---- layer 2: 64 x OUTD (N padded to 16) ------------------------------
    #pragma unroll
    for (int i = 0; i < 16; ++i) {
        int k = kb + (i < 8 ? i : i + 8);
        a0[i] = lds[wid][l15][k];
        a1[i] = lds[wid][l15][32 + k];
    }
    {
        int col = l15;
        v16h b0, b1;
        #pragma unroll
        for (int i = 0; i < 16; ++i) {
            int k = kb + (i < 8 ? i : i + 8);
            b0[i] = (col < OUTD) ? (_Float16)W2[k * OUTD + col]        : (_Float16)0.0f;
            b1[i] = (col < OUTD) ? (_Float16)W2[(32 + k) * OUTD + col] : (_Float16)0.0f;
        }
        v8f cacc = {};
        cacc = __builtin_amdgcn_wmma_f32_16x16x32_f16(false, a0, false, b0,
                                                      (short)0, cacc, false, false);
        cacc = __builtin_amdgcn_wmma_f32_16x16x32_f16(false, a1, false, b1,
                                                      (short)0, cacc, false, false);
        if (col < OUTD) {
            float bias = B2[col];
            #pragma unroll
            for (int r = 0; r < 8; ++r) {
                int atom = tile * 16 + r + 8 * half;
                Y[(size_t)atom * OUTD + col] = cacc[r] + bias;
            }
        }
    }
}

// ---------------------------------------------------------------------------
extern "C" void kernel_launch(void* const* d_in, const int* in_sizes, int n_in,
                              void* d_out, int out_size, void* d_ws, size_t ws_size,
                              hipStream_t stream)
{
    (void)in_sizes; (void)n_in; (void)out_size; (void)ws_size;

    const float* cart    = (const float*)d_in[0];   // (3, NATOM)
    const int*   aidx    = (const int*)  d_in[1];   // (2, NEIGH)
    const float* shifts  = (const float*)d_in[2];   // (3, NEIGH)
    const float* species = (const float*)d_in[3];   // (NATOM,)
    // emb params: 4..9   mp0: 10..15   mp1: 16..21 (dead)   out: 22..27
    const float* eW0 = (const float*)d_in[4],  *eB0 = (const float*)d_in[5];
    const float* eW1 = (const float*)d_in[6],  *eB1 = (const float*)d_in[7];
    const float* eW2 = (const float*)d_in[8],  *eB2 = (const float*)d_in[9];
    const float* mW0 = (const float*)d_in[10], *mB0 = (const float*)d_in[11];
    const float* mW1 = (const float*)d_in[12], *mB1 = (const float*)d_in[13];
    const float* mW2 = (const float*)d_in[14], *mB2 = (const float*)d_in[15];
    const float* oW0 = (const float*)d_in[22], *oB0 = (const float*)d_in[23];
    const float* oW1 = (const float*)d_in[24], *oB1 = (const float*)d_in[25];
    const float* oW2 = (const float*)d_in[26], *oB2 = (const float*)d_in[27];

    float* ws = (float*)d_ws;
    float* emb     = ws;  ws += (size_t)NATOM * 24;
    float* coefA   = ws;  ws += (size_t)NATOM * 8;
    float* density = ws;  ws += (size_t)NATOM * 24;
    float* bufA    = ws;  ws += (size_t)NATOM * 72;
    float* bufB    = ws;  ws += (size_t)NATOM * 72;
    float* fcE     = ws;  ws += (size_t)NEIGH;
    float* wE      = ws;  ws += (size_t)NEIGH * 8;
    float* sphE    = ws;  ws += (size_t)NEIGH * 9;

    const int AB = (NATOM + 255) / 256;   // atom blocks
    const int EB = (NEIGH + 255) / 256;   // edge blocks
    const int MB = (NATOM / 16 + 3) / 4;  // 3125 wave-tiles, 4 waves/block

    hipMemsetAsync(density, 0, (size_t)NATOM * 24 * sizeof(float), stream);
    hipMemsetAsync(bufA,    0, (size_t)NATOM * 72 * sizeof(float), stream);
    hipMemsetAsync(bufB,    0, (size_t)NATOM * 72 * sizeof(float), stream);

    emb_kernel<<<AB, 256, 0, stream>>>(species, eW0, eB0, eW1, eB1, eW2, eB2,
                                       emb, coefA);
    edge_pre_kernel<<<EB, 256, 0, stream>>>(cart, aidx, shifts, emb,
                                            fcE, wE, sphE);

    // step 0: MP_sph = 0 (bufA) -> sum_sph in bufB
    edge_scatter_kernel<<<EB, 256, 0, stream>>>(aidx, fcE, wE, sphE, coefA,
                                                bufA, bufB);
    density_kernel<<<AB, 256, 0, stream>>>(bufB, density);
    mlp_wmma_kernel<8><<<MB, 128, 0, stream>>>(density, mW0, mB0, mW1, mB1,
                                               mW2, mB2, coefA);

    // step 1: gather bufB -> scatter bufA (still zero)
    edge_scatter_kernel<<<EB, 256, 0, stream>>>(aidx, fcE, wE, sphE, coefA,
                                                bufB, bufA);
    density_kernel<<<AB, 256, 0, stream>>>(bufA, density);
    // mp_params[1] MLP output is dead in the reference -> skipped.

    // step 2 (post-loop): gather bufA -> scatter bufB (re-zeroed)
    hipMemsetAsync(bufB, 0, (size_t)NATOM * 72 * sizeof(float), stream);
    edge_scatter_kernel<<<EB, 256, 0, stream>>>(aidx, fcE, wE, sphE, coefA,
                                                bufA, bufB);
    density_kernel<<<AB, 256, 0, stream>>>(bufB, density);

    mlp_wmma_kernel<1><<<MB, 128, 0, stream>>>(density, oW0, oB0, oW1, oB1,
                                               oW2, oB2, (float*)d_out);
}
